// GATv2_40802189312612
// MI455X (gfx1250) — matrix operand order repaired and checked by reading, not compiled, not run on hardware
//
#include <hip/hip_runtime.h>
#include <hip/hip_fp16.h>

typedef __attribute__((ext_vector_type(16))) _Float16 v16h;
typedef __attribute__((ext_vector_type(8)))  _Float16 h8;
typedef __attribute__((ext_vector_type(8)))  float    v8f;
typedef __attribute__((ext_vector_type(4)))  unsigned int u32x4;
typedef __attribute__((ext_vector_type(8)))  int i32x8;
typedef __attribute__((ext_vector_type(4)))  int i32x4;

#define DD  128   // node feature dim
#define HH  2     // heads
#define CH  256   // HH*DD
#define TT  3     // edge types
#define LDSW 136  // padded LDS row stride in f16 (128 + 8; 272B -> conflict-free banks)

static __device__ __forceinline__ float leaky(float v, float s) { return v > 0.f ? v : v * s; }

// order-preserving float <-> uint encoding for atomic segment-max
static __device__ __forceinline__ unsigned encf(float f) {
    unsigned u = __float_as_uint(f);
    return (u & 0x80000000u) ? ~u : (u | 0x80000000u);
}
static __device__ __forceinline__ float decf(unsigned u) {
    return (u & 0x80000000u) ? __uint_as_float(u & 0x7fffffffu) : __uint_as_float(~u);
}

static __host__ __device__ inline int cdiv(int a, int b) { return (a + b - 1) / b; }

// ---------------------------------------------------------------------------
// f32 -> f16 conversion (activations, once per pass)
// ---------------------------------------------------------------------------
__global__ void f32_to_f16(const float* __restrict__ in, _Float16* __restrict__ out, long n) {
    long i = (long)blockIdx.x * blockDim.x + threadIdx.x;
    long st = (long)gridDim.x * blockDim.x;
    for (; i < n; i += st) out[i] = (_Float16)in[i];
}

// f32 -> f16 with per-matrix transpose: in = nmat x [DD x CH] row-major,
// out = nmat x [CH x DD] (K contiguous) -- done once, reused across passes
__global__ void f32_to_f16_T(const float* __restrict__ in, _Float16* __restrict__ out, int nmat) {
    long i = (long)blockIdx.x * blockDim.x + threadIdx.x;
    long st = (long)gridDim.x * blockDim.x;
    const long tot = (long)nmat * DD * CH;
    for (; i < tot; i += st) {
        const int  m = (int)(i / (DD * CH));
        const long r = i % (DD * CH);
        const int  c = (int)(r / DD);     // output row = weight column (0..255)
        const int  k = (int)(r % DD);     // K index (contiguous in output)
        out[i] = (_Float16)in[(size_t)m * DD * CH + (size_t)k * CH + c];
    }
}

// ---------------------------------------------------------------------------
// Y[N x 256] = X16[N x 128] @ W[128 x 256] + bias[256]
// W16T is the transposed f16 weight (256 x 128, K contiguous).
// block = 128 threads (4 waves); blockIdx.x = 16-row tile; blockIdx.y = 64-col
// group. Wave 0 TDM-loads the 64x128 f16 weight tile into LDS with hardware
// row padding (272B stride); each wave runs K=128 as 4x WMMA 16x16x32.
// ---------------------------------------------------------------------------
__global__ __launch_bounds__(128)
void gemm_xw_wmma(const _Float16* __restrict__ X16, const _Float16* __restrict__ W16T,
                  const float* __restrict__ bias, float* __restrict__ Y)
{
    __shared__ _Float16 lds_w[64 * LDSW];  // 17 KB: 64 columns x 128 K (+pad)
    const int lane    = threadIdx.x & 31;
    const int wave    = threadIdx.x >> 5;
    const int row0    = blockIdx.x * 16;
    const int colbase = blockIdx.y * 64;

    if (wave == 0) {
        // Tensor DMA descriptor (ISA 8.3/8.4): tile = 64 rows of 128 contiguous
        // f16 K-values from the 256x128 transposed weight tensor, with LDS
        // padding of 4 DWORDs after every 64 DWORDs (row stride 272B).
        const unsigned lds_off = (unsigned)(size_t)(&lds_w[0]);
        const unsigned long long ga =
            (unsigned long long)(size_t)(W16T + (size_t)colbase * DD);
        u32x4 g0;
        g0[0] = 1u;                                        // count=1 (valid user D#)
        g0[1] = lds_off;                                   // lds_addr
        g0[2] = (unsigned)(ga & 0xffffffffu);              // global_addr[31:0]
        g0[3] = (unsigned)((ga >> 32) & 0x01ffffffu)       // global_addr[56:32]
              | 0x80000000u;                               // type=2 ("image")
        i32x8 g1;
        g1[0] = (1 << 16)                                  // data_size=1 (2 bytes)
              | (1 << 20)                                  // pad_enable
              | (5 << 22)                                  // pad_interval: 64 DWORDs
              | (3 << 25);                                 // pad_amount: 4 DWORDs
        g1[1] = (int)(((unsigned)DD & 0xffffu) << 16);     // tensor_dim0 = 128 (lo16)
        g1[2] = (int)(((unsigned)DD >> 16) | ((unsigned)CH << 16)); // dim0 hi | dim1=256 lo
        g1[3] = (int)(((unsigned)CH >> 16) | ((unsigned)DD << 16)); // dim1 hi | tile_dim0=128
        g1[4] = 64;                                        // tile_dim1=64, tile_dim2=0
        g1[5] = (int)(unsigned)DD;                         // tensor_dim0_stride = 128
        g1[6] = 0;
        g1[7] = 0;
        i32x4 gz = {0, 0, 0, 0};
        i32x8 gz8 = {0, 0, 0, 0, 0, 0, 0, 0};
        __builtin_amdgcn_tensor_load_to_lds(g0, g1, gz, gz, gz8, 0);
        __builtin_amdgcn_s_wait_tensorcnt(0);
    }
    __syncthreads();

    const int half   = lane >> 4;        // K-half select
    const int lidx   = lane & 15;
    const int colloc = wave * 16 + lidx; // column within the LDS tile

    v8f c = {};
    #pragma unroll
    for (int kk = 0; kk < DD; kk += 32) {
        v16h a, b;
        // A: row = row0+lidx ; elems 0..7 -> K=kk+half*8+i ; 8..15 -> +16
        const _Float16* xrow = X16 + (size_t)(row0 + lidx) * DD + kk + half * 8;
        h8 alo = *(const h8*)xrow;
        h8 ahi = *(const h8*)(xrow + 16);
        #pragma unroll
        for (int i = 0; i < 8; ++i) { a[i] = alo[i]; a[8 + i] = ahi[i]; }
        // B from LDS: elems 0..7 -> K = kk + 8*half + j ; 8..15 -> +16, col = colloc
        const _Float16* wcol = lds_w + colloc * LDSW + kk + 8 * half;
        h8 blo = *(const h8*)wcol;
        h8 bhi = *(const h8*)(wcol + 16);
        #pragma unroll
        for (int i = 0; i < 8; ++i) { b[i] = blo[i]; b[8 + i] = bhi[i]; }

        c = __builtin_amdgcn_wmma_f32_16x16x32_f16(false, a, false, b,
                                                   (short)0, c, false, false);
    }
    const int n  = colbase + colloc;
    const float bn = bias[n];
    #pragma unroll
    for (int r = 0; r < 8; ++r)
        Y[(size_t)(row0 + r + 8 * half) * CH + n] = c[r] + bn;
}

// ---------------------------------------------------------------------------
__global__ void zero_f32(float* p, long n) {
    long i = (long)blockIdx.x * blockDim.x + threadIdx.x;
    long st = (long)gridDim.x * blockDim.x;
    for (; i < n; i += st) p[i] = 0.f;
}

// per-node init: self-loop logit, seed segment-max, zero denom/out_acc
__global__ void init_node_kernel(const float* __restrict__ xl, const float* __restrict__ xr,
                                 const float* __restrict__ attT,
                                 float* __restrict__ logit_self, unsigned* __restrict__ m_enc,
                                 float* __restrict__ denom, float* __restrict__ out_acc, int n)
{
    const int node = blockIdx.x * 8 + (threadIdx.x >> 5);
    const int lane = threadIdx.x & 31;
    if (node >= n) return;
    const size_t b = (size_t)node * CH;
    float l0 = 0.f, l1 = 0.f;
    #pragma unroll
    for (int i = 0; i < 4; ++i) {
        const int d = lane + 32 * i;
        float v0 = xl[b + d]      + xr[b + d];
        float v1 = xl[b + DD + d] + xr[b + DD + d];
        v0 = leaky(v0, 0.2f); v1 = leaky(v1, 0.2f);
        l0 += v0 * attT[d];
        l1 += v1 * attT[DD + d];
    }
    #pragma unroll
    for (int off = 16; off >= 1; off >>= 1) {
        l0 += __shfl_xor(l0, off);
        l1 += __shfl_xor(l1, off);
    }
    if (lane == 0) {
        logit_self[node * 2 + 0] = l0;
        logit_self[node * 2 + 1] = l1;
        m_enc[node * 2 + 0] = encf(l0);
        m_enc[node * 2 + 1] = encf(l1);
        denom[node * 2 + 0] = 0.f;
        denom[node * 2 + 1] = 0.f;
    }
    #pragma unroll
    for (int i = 0; i < 8; ++i) out_acc[b + lane + 32 * i] = 0.f;
}

// per-edge logits + atomic segment max (thread per edge)
__global__ void edge_logit_kernel(const int* __restrict__ src, const int* __restrict__ dst,
                                  const int* __restrict__ eattr, int t,
                                  const float* __restrict__ xl, const float* __restrict__ xr,
                                  const float* __restrict__ attT,
                                  float* __restrict__ logitE, unsigned* __restrict__ m_enc, int E_)
{
    const int e = blockIdx.x * blockDim.x + threadIdx.x;
    if (e >= E_ || eattr[e] != t) return;
    const size_t sb = (size_t)src[e] * CH;
    const size_t db = (size_t)dst[e] * CH;
    float l0 = 0.f, l1 = 0.f;
    for (int d = 0; d < DD; ++d) {
        float v0 = leaky(xl[sb + d]      + xr[db + d],      0.2f);
        float v1 = leaky(xl[sb + DD + d] + xr[db + DD + d], 0.2f);
        l0 += v0 * attT[d];
        l1 += v1 * attT[DD + d];
    }
    logitE[(size_t)e * 2 + 0] = l0;
    logitE[(size_t)e * 2 + 1] = l1;
    const int dn = dst[e];
    atomicMax(&m_enc[dn * 2 + 0], encf(l0));
    atomicMax(&m_enc[dn * 2 + 1], encf(l1));
}

// softmax numerator + message scatter (wave per edge)
__global__ void edge_accum_kernel(const int* __restrict__ src, const int* __restrict__ dst,
                                  const int* __restrict__ eattr, int t,
                                  const float* __restrict__ xl, const float* __restrict__ logitE,
                                  const unsigned* __restrict__ m_enc,
                                  float* __restrict__ denom, float* __restrict__ out_acc, int E_)
{
    const int e    = blockIdx.x * 8 + (threadIdx.x >> 5);
    const int lane = threadIdx.x & 31;
    if (e >= E_ || eattr[e] != t) return;
    const int s = src[e], dn = dst[e];
    #pragma unroll
    for (int h = 0; h < HH; ++h) {
        const float m = decf(m_enc[dn * 2 + h]);
        const float p = __expf(logitE[(size_t)e * 2 + h] - m);
        if (lane == 0) atomicAdd(&denom[dn * 2 + h], p);
        const float* xs = xl + (size_t)s * CH + h * DD;
        float* oa = out_acc + (size_t)dn * CH + h * DD;
        __builtin_prefetch(xs, 0, 0);
        #pragma unroll
        for (int i = 0; i < 4; ++i) {
            const int d = lane + 32 * i;
            atomicAdd(&oa[d], p * xs[d]);
        }
    }
}

// add self-loop contribution, divide, head-mean, bias, leaky -> acc
__global__ void finalize_kernel(const float* __restrict__ xl, const float* __restrict__ out_acc,
                                const float* __restrict__ denom, const float* __restrict__ logit_self,
                                const unsigned* __restrict__ m_enc, const float* __restrict__ biasT,
                                float* __restrict__ acc, int n)
{
    const int node = blockIdx.x;
    const int d    = threadIdx.x;
    if (node >= n) return;
    const float ps0 = __expf(logit_self[node * 2 + 0] - decf(m_enc[node * 2 + 0]));
    const float ps1 = __expf(logit_self[node * 2 + 1] - decf(m_enc[node * 2 + 1]));
    const float dn0 = denom[node * 2 + 0] + ps0;
    const float dn1 = denom[node * 2 + 1] + ps1;
    const size_t b  = (size_t)node * CH;
    const float v0  = (out_acc[b + d]      + ps0 * xl[b + d])      / dn0;
    const float v1  = (out_acc[b + DD + d] + ps1 * xl[b + DD + d]) / dn1;
    const float h   = 0.5f * (v0 + v1) + biasT[d];
    acc[(size_t)node * DD + d] += leaky(h, 0.01f);
}

__global__ void scale_kernel(const float* __restrict__ a, float* __restrict__ o, long n, float s) {
    long i = (long)blockIdx.x * blockDim.x + threadIdx.x;
    long st = (long)gridDim.x * blockDim.x;
    for (; i < n; i += st) o[i] = a[i] * s;
}

__global__ void pool_kernel(const float* __restrict__ x, const int* __restrict__ batch,
                            float* __restrict__ pooled, int n)
{
    long i = (long)blockIdx.x * blockDim.x + threadIdx.x;
    long st = (long)gridDim.x * blockDim.x;
    const long tot = (long)n * DD;
    for (; i < tot; i += st) {
        const int node = (int)(i >> 7);
        const int d    = (int)(i & (DD - 1));
        atomicAdd(&pooled[(size_t)batch[node] * DD + d], x[i]);
    }
}

// tiny MLP head: one block per graph, block=96
__global__ void mlp_kernel(const float* __restrict__ pooled, const float* __restrict__ pt,
                           const float* __restrict__ W1, const float* __restrict__ b1,
                           const float* __restrict__ W2, const float* __restrict__ b2,
                           const float* __restrict__ W3, const float* __restrict__ b3,
                           float* __restrict__ out, int OUTD)
{
    __shared__ float h1[80];
    __shared__ float h2[80];
    const int g = blockIdx.x;
    const int j = threadIdx.x;
    if (j < 80) {
        float s = b1[j];
        for (int k = 0; k < DD; ++k) s += pooled[(size_t)g * DD + k] * W1[k * 80 + j];
        s += pt[g] * W1[DD * 80 + j];
        h1[j] = leaky(s, 0.01f);
    }
    __syncthreads();
    if (j < 80) {
        float s = b2[j];
        for (int k = 0; k < 80; ++k) s += h1[k] * W2[k * 80 + j];
        h2[j] = leaky(s, 0.01f);
    }
    __syncthreads();
    if (j < OUTD) {
        float s = b3[j];
        for (int k = 0; k < 80; ++k) s += h2[k] * W3[k * OUTD + j];
        out[g * OUTD + j] = s;
    }
}

// ---------------------------------------------------------------------------
extern "C" void kernel_launch(void* const* d_in, const int* in_sizes, int n_in,
                              void* d_out, int out_size, void* d_ws, size_t ws_size,
                              hipStream_t stream)
{
    (void)n_in; (void)ws_size;
    const float* X     = (const float*)d_in[0];
    const int*   eidx  = (const int*)  d_in[1];
    const int*   eattr = (const int*)  d_in[2];
    const int*   batch = (const int*)  d_in[3];
    const float* pType = (const float*)d_in[4];
    const float* Wl    = (const float*)d_in[5];
    const float* bl    = (const float*)d_in[6];
    const float* Wr    = (const float*)d_in[7];
    const float* br    = (const float*)d_in[8];
    const float* att   = (const float*)d_in[9];
    const float* biasg = (const float*)d_in[10];
    const float* W1    = (const float*)d_in[11];
    const float* b1    = (const float*)d_in[12];
    const float* W2    = (const float*)d_in[13];
    const float* b2    = (const float*)d_in[14];
    const float* W3    = (const float*)d_in[15];
    const float* b3    = (const float*)d_in[16];

    const int Nn = in_sizes[0] / DD;
    const int Ee = in_sizes[2];
    const int Gg = in_sizes[4];
    const int OUTD = out_size / Gg;
    const int* src = eidx;
    const int* dst = eidx + Ee;

    // workspace partition
    float* ws = (float*)d_ws;
    float*    xl         = ws;                          // N*256
    float*    xr         = xl + (size_t)Nn * CH;        // N*256
    float*    out_acc    = xr + (size_t)Nn * CH;        // N*256
    float*    acc        = out_acc + (size_t)Nn * CH;   // N*128
    float*    xcur       = acc + (size_t)Nn * DD;       // N*128
    float*    logitE     = xcur + (size_t)Nn * DD;      // E*2
    float*    logit_self = logitE + (size_t)Ee * 2;     // N*2
    float*    denom      = logit_self + (size_t)Nn * 2; // N*2
    unsigned* m_enc      = (unsigned*)(denom + (size_t)Nn * 2); // N*2
    float*    pooled     = (float*)(m_enc + (size_t)Nn * 2);    // G*128
    _Float16* X16        = (_Float16*)(pooled + (size_t)Gg * DD); // N*128 f16
    _Float16* Wl16T      = X16 + (size_t)Nn * DD;                 // T*256*128 f16 (transposed)
    _Float16* Wr16T      = Wl16T + (size_t)TT * DD * CH;          // T*256*128 f16 (transposed)

    const dim3 gemmGrid(Nn / 16, 4);
    const int  zgrid = 2048;

    // one-time transposed f16 weight conversion (reused across passes)
    f32_to_f16_T<<<cdiv(TT * DD * CH, 256), 256, 0, stream>>>(Wl, Wl16T, TT);
    f32_to_f16_T<<<cdiv(TT * DD * CH, 256), 256, 0, stream>>>(Wr, Wr16T, TT);

    const float* xin = X;
    for (int pass = 0; pass < 2; ++pass) {
        f32_to_f16<<<zgrid, 256, 0, stream>>>(xin, X16, (long)Nn * DD);
        zero_f32<<<zgrid, 256, 0, stream>>>(acc, (long)Nn * DD);
        for (int t = 0; t < TT; ++t) {
            const float* attT = att + (size_t)t * HH * DD;
            gemm_xw_wmma<<<gemmGrid, 128, 0, stream>>>(X16, Wl16T + (size_t)t * DD * CH,
                                                       bl + (size_t)t * CH, xl);
            gemm_xw_wmma<<<gemmGrid, 128, 0, stream>>>(X16, Wr16T + (size_t)t * DD * CH,
                                                       br + (size_t)t * CH, xr);
            init_node_kernel<<<cdiv(Nn, 8), 256, 0, stream>>>(xl, xr, attT, logit_self,
                                                              m_enc, denom, out_acc, Nn);
            edge_logit_kernel<<<cdiv(Ee, 256), 256, 0, stream>>>(src, dst, eattr, t, xl, xr,
                                                                 attT, logitE, m_enc, Ee);
            edge_accum_kernel<<<cdiv(Ee, 8), 256, 0, stream>>>(src, dst, eattr, t, xl, logitE,
                                                               m_enc, denom, out_acc, Ee);
            finalize_kernel<<<Nn, DD, 0, stream>>>(xl, out_acc, denom, logit_self, m_enc,
                                                   biasg + (size_t)t * DD, acc, Nn);
        }
        scale_kernel<<<zgrid, 256, 0, stream>>>(acc, xcur, (long)Nn * DD, 1.f / 3.f);
        xin = xcur;
    }

    zero_f32<<<cdiv(Gg * DD, 256), 256, 0, stream>>>(pooled, (long)Gg * DD);
    pool_kernel<<<zgrid, 256, 0, stream>>>(xin, batch, pooled, Nn);
    mlp_kernel<<<Gg, 96, 0, stream>>>(pooled, pType, W1, b1, W2, b2, W3, b3,
                                      (float*)d_out, OUTD);
}
